// GraphWeightEstimator_50878182588739
// MI455X (gfx1250) — matrix operand order repaired and checked by reading, not compiled, not run on hardware
//
#include <hip/hip_runtime.h>
#include <math.h>

typedef _Float16 v8h  __attribute__((ext_vector_type(8)));
typedef _Float16 v16h __attribute__((ext_vector_type(16)));
typedef float    v8f  __attribute__((ext_vector_type(8)));

#define HW    400
#define NPIX  (HW*HW)
#define CKK   363        // 3*11*11 unfolded K
#define N1    128
#define N2    64
#define NE    8
#define K1CH  12         // 384 / 32 k-chunks for layer 1 (padded)
#define K2CH  4          // 128 / 32 k-chunks for layer 2
#define NT1   8          // 128 / 16 n-tiles
#define NT2   4          // 64 / 16 n-tiles
#define NW    4          // waves per block (128 threads)

#define PW 26            // patch width  (16 + 10 halo)
#define PH 18            // patch height (8 + 10 halo)
#define PCH 480          // per-channel patch stride (halves)
#define PATCH_MAIN (3*PCH)          // 1440
#define PATCH_TOT  (PATCH_MAIN+224) // + zero guard for K-padding reads

#define W1F_HALVES (K1CH*NT1*32*16) // 49152
#define W2F_HALVES (K2CH*NT2*32*16) // 8192

// ---- Pack W1 (363x128 f32, row-major [k][n]) into f16 WMMA B-fragments ----
// frag layout: [kc][nt][lane][16 halves]; B 32x16 f16 tile:
//   lanes 0-15 hold K=0..7,16..23, lanes 16-31 hold K=8..15,24..31, N = lane&15
__global__ __launch_bounds__(256) void prep_w1_k(const float* __restrict__ W1,
                                                 _Float16* __restrict__ W1f) {
  int t = blockIdx.x * 256 + threadIdx.x;
  if (t >= W1F_HALVES) return;
  int h = t & 15, lane = (t >> 4) & 31, nt = (t >> 9) & 7, kc = t >> 12;
  int k = kc * 32 + ((lane & 16) >> 1) + (h < 8 ? h : 8 + h);
  int n = nt * 16 + (lane & 15);
  float v = (k < CKK) ? W1[k * N1 + n] : 0.f;
  W1f[t] = (_Float16)v;
}

__global__ __launch_bounds__(256) void prep_w2_k(const float* __restrict__ W2,
                                                 _Float16* __restrict__ W2f) {
  int t = blockIdx.x * 256 + threadIdx.x;
  if (t >= W2F_HALVES) return;
  int h = t & 15, lane = (t >> 4) & 31, nt = (t >> 9) & 3, kc = t >> 11;
  int k = kc * 32 + ((lane & 16) >> 1) + (h < 8 ? h : 8 + h);     // K = 128 exactly, no padding
  int n = nt * 16 + (lane & 15);
  W2f[t] = (_Float16)W2[k * N2 + n];
}

// ---- Fused unfold + MLP + normalize ----
// Block = 4 waves = 128 threads, tile = 8 rows x 16 cols of pixels.
// Each wave owns TWO image rows (32 pixels, two A-fragment sets) so every
// B-fragment load feeds two WMMAs; B fragments are double-buffered so the
// next global load is in flight while the current pair of WMMAs executes.
__global__ __launch_bounds__(128) void conv_mlp_k(
    const float* __restrict__ x,
    const float* __restrict__ b1, const float* __restrict__ b2,
    const float* __restrict__ W3, const float* __restrict__ b3,
    const _Float16* __restrict__ W1f, const _Float16* __restrict__ W2f,
    float* __restrict__ out)
{
  __shared__ alignas(32) _Float16 patch[PATCH_TOT];
  __shared__ unsigned short offt[384];
  __shared__ alignas(32) _Float16 h1[NW][32 * N1];   // per-wave [pix][ch]
  __shared__ alignas(32) _Float16 h2[NW][32 * N2];   // per-wave [pix][ch]
  __shared__ float b1s[N1];
  __shared__ float b2s[N2];
  __shared__ float w3s[N2 * NE];
  __shared__ float b3s[NE];

  const int tid = threadIdx.x;
  const int y0 = blockIdx.y * 8;
  const int x0 = blockIdx.x * 16;

  // Stage input patch (f32 -> f16) with zero border padding + zero guard.
  for (int idx = tid; idx < PATCH_TOT; idx += 128) {
    float v = 0.f;
    if (idx < PATCH_MAIN) {
      int c = idx / PCH, r = idx - c * PCH;
      int pr = r / PW, pc = r - pr * PW;
      if (pr < PH) {
        int gy = y0 - 5 + pr, gx = x0 - 5 + pc;
        if ((unsigned)gy < (unsigned)HW && (unsigned)gx < (unsigned)HW)
          v = x[c * NPIX + gy * HW + gx];
      }
    }
    patch[idx] = (_Float16)v;
  }
  // Unfold-k -> patch offset table (channel slowest, (ki,kj) fastest).
  for (int i = tid; i < 384; i += 128) {
    unsigned short off;
    if (i < CKK) {
      int c = i / 121, rem = i - c * 121;
      int ki = rem / 11, kj = rem - ki * 11;
      off = (unsigned short)(c * PCH + ki * PW + kj);
    } else {
      off = (unsigned short)PATCH_MAIN;   // points into zero guard
    }
    offt[i] = off;
  }
  if (tid < N1) b1s[tid] = b1[tid];
  if (tid < N2) b2s[tid] = b2[tid];
  for (int i = tid; i < N2 * NE; i += 128) w3s[i] = W3[i];
  if (tid < NE) b3s[tid] = b3[tid];
  __syncthreads();

  const int lane = tid & 31;
  const int w = tid >> 5;              // wave id: owns image rows 2w, 2w+1
  const int m = lane & 15;             // pixel column within tile
  const int hiHalf = (lane >> 4) & 1;
  const int baseIdx0 = (2 * w)     * PW + m;   // row 2w   pixel origin
  const int baseIdx1 = (2 * w + 1) * PW + m;   // row 2w+1 pixel origin
  _Float16* h1w = &h1[w][0];
  _Float16* h2w = &h2[w][0];

  const v8f vzero = {0.f, 0.f, 0.f, 0.f, 0.f, 0.f, 0.f, 0.f};
  const int Mb = hiHalf * 8;

  // ---------------- Layer 1: two [16 x 384] @ [384 x 128] ----------------
  v8f acc[2][NT1];
  #pragma unroll
  for (int nt = 0; nt < NT1; ++nt) { acc[0][nt] = vzero; acc[1][nt] = vzero; }

  // Double-buffered B fragment stream (frag idx = kc*NT1 + nt).
  v16h bnext = *(const v16h*)(W1f + (0 * 32 + lane) * 16);

  #pragma unroll
  for (int kc = 0; kc < K1CH; ++kc) {
    const int kb = kc * 32 + hiHalf * 8;
    v16h a0, a1;
    // A 16x32 f16 layout: lane<16 -> K kb..kb+7, kb+16..kb+23 ; lane>=16 +8
    #pragma unroll
    for (int h = 0; h < 16; ++h) {
      int k = kb + (h < 8 ? h : 8 + h);
      int off = (int)offt[k];
      a0[h] = patch[baseIdx0 + off];
      a1[h] = patch[baseIdx1 + off];
    }
    #pragma unroll
    for (int nt = 0; nt < NT1; ++nt) {
      v16h bcur = bnext;
      const int nidx = kc * NT1 + nt + 1;
      if (nidx < K1CH * NT1)
        bnext = *(const v16h*)(W1f + (nidx * 32 + lane) * 16);
      acc[0][nt] = __builtin_amdgcn_wmma_f32_16x16x32_f16(
          false, a0, false, bcur, (short)0, acc[0][nt], false, false);
      acc[1][nt] = __builtin_amdgcn_wmma_f32_16x16x32_f16(
          false, a1, false, bcur, (short)0, acc[1][nt], false, false);
    }
  }

  // Bias + leaky ReLU (0.01), spill to LDS as f16 [pix][ch].
  #pragma unroll
  for (int s = 0; s < 2; ++s) {
    #pragma unroll
    for (int nt = 0; nt < NT1; ++nt) {
      float bias = b1s[nt * 16 + m];
      #pragma unroll
      for (int r = 0; r < 8; ++r) {
        float v = acc[s][nt][r] + bias;
        v = fmaxf(v, 0.f) + 0.01f * fminf(v, 0.f);
        h1w[(s * 16 + Mb + r) * N1 + nt * 16 + m] = (_Float16)v;
      }
    }
  }

  // ---------------- Layer 2: two [16 x 128] @ [128 x 64] ----------------
  v8f acc2[2][NT2];
  #pragma unroll
  for (int nt = 0; nt < NT2; ++nt) { acc2[0][nt] = vzero; acc2[1][nt] = vzero; }

  v16h bnext2 = *(const v16h*)(W2f + (0 * 32 + lane) * 16);

  #pragma unroll
  for (int kc = 0; kc < K2CH; ++kc) {
    const int kb = kc * 32 + hiHalf * 8;
    v16h a2[2];
    #pragma unroll
    for (int s = 0; s < 2; ++s) {
      v8h lo = *(const v8h*)(h1w + (s * 16 + m) * N1 + kb);        // ds_load_b128
      v8h hi = *(const v8h*)(h1w + (s * 16 + m) * N1 + kb + 16);   // ds_load_b128
      #pragma unroll
      for (int h = 0; h < 8; ++h) { a2[s][h] = lo[h]; a2[s][h + 8] = hi[h]; }
    }
    #pragma unroll
    for (int nt = 0; nt < NT2; ++nt) {
      v16h bcur = bnext2;
      const int nidx = kc * NT2 + nt + 1;
      if (nidx < K2CH * NT2)
        bnext2 = *(const v16h*)(W2f + (nidx * 32 + lane) * 16);
      acc2[0][nt] = __builtin_amdgcn_wmma_f32_16x16x32_f16(
          false, a2[0], false, bcur, (short)0, acc2[0][nt], false, false);
      acc2[1][nt] = __builtin_amdgcn_wmma_f32_16x16x32_f16(
          false, a2[1], false, bcur, (short)0, acc2[1][nt], false, false);
    }
  }

  #pragma unroll
  for (int s = 0; s < 2; ++s) {
    #pragma unroll
    for (int nt = 0; nt < NT2; ++nt) {
      float bias = b2s[nt * 16 + m];
      #pragma unroll
      for (int r = 0; r < 8; ++r) {
        float v = acc2[s][nt][r] + bias;
        v = fmaxf(v, 0.f) + 0.01f * fminf(v, 0.f);
        h2w[(s * 16 + Mb + r) * N2 + nt * 16 + m] = (_Float16)v;
      }
    }
  }

  // ---------------- Layer 3 (64 -> 8) + L2 normalize: one pixel per lane ----
  {
    const int p = lane;                  // local pixel 0..31
    float o[NE];
    #pragma unroll
    for (int e = 0; e < NE; ++e) o[e] = b3s[e];
    #pragma unroll 8
    for (int j = 0; j < N2; ++j) {
      float hv = (float)h2w[p * N2 + j];
      #pragma unroll
      for (int e = 0; e < NE; ++e) o[e] = fmaf(hv, w3s[j * NE + e], o[e]);
    }
    float ss = 0.f;
    #pragma unroll
    for (int e = 0; e < NE; ++e) ss += o[e] * o[e];
    float inv = 1.f / fmaxf(sqrtf(ss), 1e-12f);
    const int gy = y0 + 2 * w + (p >> 4);
    const int gx = x0 + (p & 15);
    float* op = out + (gy * HW + gx) * NE;
    float4 p0 = { o[0] * inv, o[1] * inv, o[2] * inv, o[3] * inv };
    float4 p1 = { o[4] * inv, o[5] * inv, o[6] * inv, o[7] * inv };
    *(float4*)op = p0;
    *(float4*)(op + 4) = p1;
  }
}

extern "C" void kernel_launch(void* const* d_in, const int* in_sizes, int n_in,
                              void* d_out, int out_size, void* d_ws, size_t ws_size,
                              hipStream_t stream) {
  (void)in_sizes; (void)n_in; (void)out_size; (void)ws_size;
  const float* x  = (const float*)d_in[0];
  const float* W1 = (const float*)d_in[1];
  const float* b1 = (const float*)d_in[2];
  const float* W2 = (const float*)d_in[3];
  const float* b2 = (const float*)d_in[4];
  const float* W3 = (const float*)d_in[5];
  const float* b3 = (const float*)d_in[6];

  _Float16* W1f = (_Float16*)d_ws;
  _Float16* W2f = (_Float16*)((char*)d_ws + (size_t)W1F_HALVES * sizeof(_Float16));

  prep_w1_k<<<(W1F_HALVES + 255) / 256, 256, 0, stream>>>(W1, W1f);
  prep_w2_k<<<(W2F_HALVES + 255) / 256, 256, 0, stream>>>(W2, W2f);
  // 25 x-tiles of 16 cols, 50 y-tiles of 8 rows -> 1250 workgroups of 4 waves
  conv_mlp_k<<<dim3(25, 50), 128, 0, stream>>>(x, b1, b2, W3, b3, W1f, W2f,
                                               (float*)d_out);
}
